// NAIS_distance_Embedding_77395310674424
// MI455X (gfx1250) — matrix operand order repaired and checked by reading, not compiled, not run on hardware
//
#include <hip/hip_runtime.h>
#include <hip/hip_bf16.h>

typedef __attribute__((ext_vector_type(16))) __bf16 v16bf;
typedef __attribute__((ext_vector_type(8)))  __bf16 v8bf;
typedef __attribute__((ext_vector_type(8)))  float  v8f;

#define B_SZ   2048
#define H_SZ   200
#define EMBED  128
#define HIDDEN 256
#define KPAD   160   // 130 padded up to 5*32 for the bf16 WMMA K dimension
#define TILES  13    // ceil(200/16)

// Transpose W1 [130,256] f32 -> w1t [256][160] bf16 (zero-padded K), so each
// lane's WMMA B-fragment (col N, 16 contiguous K) is one 32-byte load.
__global__ void prep_w1t(const float* __restrict__ W1, __bf16* __restrict__ w1t) {
  int i = blockIdx.x * blockDim.x + threadIdx.x;
  if (i >= HIDDEN * KPAD) return;
  int n = i / KPAD, k = i % KPAD;
  float v = (k < EMBED + 2) ? W1[k * HIDDEN + n] : 0.0f;
  w1t[i] = (__bf16)v;
}

__global__ __launch_bounds__(256) void nais_kernel(
    const int* __restrict__ history, const int* __restrict__ target,
    const float* __restrict__ tdist, const float* __restrict__ embH,
    const float* __restrict__ embT,  const float* __restrict__ b1,
    const float* __restrict__ W2,    const float* __restrict__ Wd,
    const float* __restrict__ bd,    const __bf16* __restrict__ w1t,
    float* __restrict__ out) {
  __shared__ float t_sm[EMBED];
  __shared__ __attribute__((aligned(32))) __bf16 a_sm[16 * KPAD];
  __shared__ float s_sm[16];
  __shared__ float attn_sm[16];
  __shared__ float mask_sm[16];
  __shared__ float acc_sm[2];

  const int tid = threadIdx.x;
  const int b   = blockIdx.x;
  const int tgt = target[b];

  if (tid < EMBED) t_sm[tid] = embT[(size_t)tgt * EMBED + tid];
  if (tid < 2)     acc_sm[tid] = 0.0f;

  const int lane = tid & 31;
  const int wave = tid >> 5;
  const int half = lane >> 4;   // which 16-lane half
  const int hl   = lane & 15;   // lane within half

  // Per-lane epilogue constants (column = N index owned by this lane)
  const int ncol0 = wave * 32 + hl;
  const int ncol1 = ncol0 + 16;
  const float b1v0 = b1[ncol0], b1v1 = b1[ncol1];
  const float w2v0 = W2[ncol0], w2v1 = W2[ncol1];
  const float wd00 = Wd[0], wd01 = Wd[1], wd10 = Wd[2], wd11 = Wd[3];
  const float bd0 = bd[0], bd1 = bd[1];

  // Hoist B fragments: invariant across all 13 tiles. contiguous B k order used here (A matches):
  // lanes 0-15 hold K=ks*32+0..15 at col N, lanes 16-31 hold K=ks*32+16..31.
  v16bf bfrag[5][2];
#pragma unroll
  for (int ks = 0; ks < 5; ++ks) {
    bfrag[ks][0] = *(const v16bf*)(w1t + (size_t)ncol0 * KPAD + ks * 32 + half * 16);
    bfrag[ks][1] = *(const v16bf*)(w1t + (size_t)ncol1 * KPAD + ks * 32 + half * 16);
  }

  const int r  = tid >> 4;  // row within tile, 0..15
  const int kg = tid & 15;  // 8-dim chunk of the embedding, 0..15

  for (int tile = 0; tile < TILES; ++tile) {
    __syncthreads();                                 // prev tile fully consumed
    if (tid < 16) { s_sm[tid] = 0.0f; attn_sm[tid] = 0.0f; }
    __syncthreads();

    // ---- Phase 1: build A tile (bf16) + row dot-sums + mask + dist ----
    const int  h     = tile * 16 + r;
    const bool valid = (h < H_SZ);
    const int  idx   = valid ? history[b * H_SZ + h] : 0;
    float partial = 0.0f;
    if (valid) {
      const float* hp = embH + (size_t)idx * EMBED + kg * 8;
#pragma unroll
      for (int i = 0; i < 8; ++i) {
        float p = hp[i] * t_sm[kg * 8 + i];
        a_sm[r * KPAD + kg * 8 + i] = (__bf16)p;
        partial += p;
      }
      atomicAdd(&s_sm[r], partial);
    } else {
#pragma unroll
      for (int i = 0; i < 8; ++i) a_sm[r * KPAD + kg * 8 + i] = (__bf16)0.0f;
    }
    if (kg == 0) {
      float d0 = 0.0f, d1 = 0.0f;
      if (valid) {
        const float td0 = tdist[(size_t)(b * H_SZ + h) * 2 + 0];
        const float td1 = tdist[(size_t)(b * H_SZ + h) * 2 + 1];
        float z0 = 1000.0f * (td0 * wd00 + td1 * wd10) + bd0;
        float z1 = 1000.0f * (td0 * wd01 + td1 * wd11) + bd1;
        d0 = 1.0f / (1.0f + expf(-z0));
        d1 = 1.0f / (1.0f + expf(-z1));
      }
      a_sm[r * KPAD + 128] = (__bf16)d0;
      a_sm[r * KPAD + 129] = (__bf16)d1;
      mask_sm[r] = (valid && idx != tgt) ? 1.0f : 0.0f;
    }
    if (kg == 1) {
      for (int k = 130; k < KPAD; ++k) a_sm[r * KPAD + k] = (__bf16)0.0f;
    }
    __syncthreads();

    // ---- Phase 2: WMMA — each wave owns 32 hidden columns ----
    // Preload ALL K-step A-fragments first so the 10 ds_load_b128 are in
    // flight together, then run the WMMA chain back-to-back (accumulator
    // chaining on C has no D->A/B hazard).
    v16bf afrag[5];
#pragma unroll
    for (int ks = 0; ks < 5; ++ks) {
      // Contiguous A k order used here (same as B): lane m=hl, K 0..15 (half 0) or
      // 16..31 (half 1) -> two contiguous 16B LDS chunks.
      const __bf16* ap = a_sm + hl * KPAD + ks * 32 + half * 16;
      v8bf lo = *(const v8bf*)ap;
      v8bf hi = *(const v8bf*)(ap + 8);
#pragma unroll
      for (int i = 0; i < 8; ++i) { afrag[ks][i] = lo[i]; afrag[ks][i + 8] = hi[i]; }
    }
    v8f c0 = {}; v8f c1 = {};
#pragma unroll
    for (int ks = 0; ks < 5; ++ks) {
      c0 = __builtin_amdgcn_wmma_f32_16x16x32_bf16(false, afrag[ks], false,
              bfrag[ks][0], (short)0, c0, false, false);
      c1 = __builtin_amdgcn_wmma_f32_16x16x32_bf16(false, afrag[ks], false,
              bfrag[ks][1], (short)0, c1, false, false);
    }

    // Epilogue: bias + relu + *W2; fire-and-forget LDS float atomics per row
    // (no cross-lane shuffle chains, no per-op waits).
    // C layout: VGPR v -> row v (lanes 0-15) / v+8 (lanes 16-31); col = hl.
#pragma unroll
    for (int v = 0; v < 8; ++v) {
      float x0 = c0[v] + b1v0; x0 = x0 > 0.0f ? x0 : 0.0f;
      float x1 = c1[v] + b1v1; x1 = x1 > 0.0f ? x1 : 0.0f;
      atomicAdd(&attn_sm[v + 8 * half], x0 * w2v0 + x1 * w2v1);
    }
    __syncthreads();

    // ---- Phase 3: exp/mask and accumulate numerator/denominator ----
    if (tid < 16) {
      float e = expf(attn_sm[tid]) * mask_sm[tid];
      atomicAdd(&acc_sm[0], e * s_sm[tid]);
      atomicAdd(&acc_sm[1], e);
    }
  }

  __syncthreads();
  if (tid == 0) {
    float pred = acc_sm[0] / sqrtf(acc_sm[1]);   // BETA = 0.5
    out[b] = 1.0f / (1.0f + expf(-pred));
  }
}

extern "C" void kernel_launch(void* const* d_in, const int* in_sizes, int n_in,
                              void* d_out, int out_size, void* d_ws, size_t ws_size,
                              hipStream_t stream) {
  const int*   history = (const int*)  d_in[0];
  const int*   target  = (const int*)  d_in[1];
  // d_in[2] history_region, d_in[3] target_region: unused by the reference
  const float* tdist   = (const float*)d_in[4];
  const float* embH    = (const float*)d_in[5];
  const float* embT    = (const float*)d_in[6];
  const float* W1      = (const float*)d_in[7];
  const float* b1      = (const float*)d_in[8];
  const float* W2      = (const float*)d_in[9];
  const float* Wd      = (const float*)d_in[10];
  const float* bd      = (const float*)d_in[11];

  __bf16* w1t = (__bf16*)d_ws;   // 256*160*2 = 80 KB of workspace

  prep_w1t<<<(HIDDEN * KPAD + 255) / 256, 256, 0, stream>>>(W1, w1t);
  nais_kernel<<<B_SZ, 256, 0, stream>>>(history, target, tdist, embH, embT,
                                        b1, W2, Wd, bd, w1t, (float*)d_out);
}
